// GraphEncoder_70454643523878
// MI455X (gfx1250) — matrix operand order repaired and checked by reading, not compiled, not run on hardware
//
#include <hip/hip_runtime.h>
#include <hip/hip_bf16.h>

typedef __attribute__((ext_vector_type(16))) __bf16 v16bf;
typedef __attribute__((ext_vector_type(8)))  __bf16 v8bf;
typedef __attribute__((ext_vector_type(8)))  float  v8f;

#define HDIM 128
#define LDP  136   // padded LDS row stride in bf16 elems (272B -> bank-spread)

// ---------------- agg init: agg = h (the "+x_i" term of GIN) ----------------
__global__ void copy_kernel(float* __restrict__ dst, const float* __restrict__ src, int n4) {
    int t = blockIdx.x * 256 + threadIdx.x;
    if (t < n4) ((float4*)dst)[t] = ((const float4*)src)[t];
}

// ---------------- edge scatter: agg[dst] += h[src], one wave per edge --------
__global__ void scatter_kernel(float* __restrict__ agg, const float* __restrict__ h,
                               const int* __restrict__ src, const int* __restrict__ dst, int E) {
    long long tid = (long long)blockIdx.x * 256 + threadIdx.x;
    long long e = tid >> 5;
    if (e >= E) return;
    int c = (int)(tid & 31) << 2;              // 4 floats per lane, 32 lanes = 128
    long long s = (long long)src[e] * HDIM + c;
    long long d = (long long)dst[e] * HDIM + c;
    const float4 v = *(const float4*)(h + s);
    float* a = agg + d;
    atomicAdd(a + 0, v.x);
    atomicAdd(a + 1, v.y);
    atomicAdd(a + 2, v.z);
    atomicAdd(a + 3, v.w);
}

// ------------- fused MLP: hout = relu(hin@W1+b1)@W2 + b2 (WMMA bf16) ---------
__global__ __launch_bounds__(256) void mlp_kernel(
    float* __restrict__ hout, const float* __restrict__ hin,
    const float* __restrict__ W1, const float* __restrict__ b1,
    const float* __restrict__ W2, const float* __restrict__ b2, int nrows)
{
    __shared__ __bf16 sW1t[HDIM * LDP];   // transposed: sW1t[n*LDP + k] = W1[k,n]
    __shared__ __bf16 sW2t[HDIM * LDP];
    __shared__ __bf16 sY[8 * 16 * LDP];   // per-wave Y staging (16 rows x 128 K)

    const int tid = threadIdx.x;
    for (int i = tid; i < HDIM * HDIM; i += 256) {
        int k = i >> 7, nc = i & 127;
        sW1t[nc * LDP + k] = (__bf16)W1[i];
        sW2t[nc * LDP + k] = (__bf16)W2[i];
    }
    __syncthreads();

    const int wave = tid >> 5;
    const int lane = tid & 31;
    const int half = lane >> 4;       // K-group selector per ISA A/B layouts
    const int l16  = lane & 15;
    const long long rowBase = (long long)blockIdx.x * 128 + wave * 16;
    const long long row = rowBase + l16;             // A-fragment row (M = lane%16)
    // Clamp instead of predicate: out-of-range rows only feed out-of-range
    // output rows (rows are independent through both GEMMs), which are masked
    // at the final store. Clamped reads stay in-bounds and branch-free.
    const long long crow = row < nrows ? row : (long long)(nrows - 1);
    const float* hbase = hin + crow * HDIM;
    __bf16* myY = sY + wave * 16 * LDP;

    v8f acc[8];
    v8f vzero = {};
    #pragma unroll
    for (int ct = 0; ct < 8; ++ct) acc[ct] = vzero;

    // ---- GEMM1: acc = A(hin) @ W1 ----
    #pragma unroll
    for (int kb = 0; kb < 4; ++kb) {                 // K blocks of 32
        // A elem j (j<8):  K = kb*32 + 8*half + j
        // A elem j (j>=8): K = kb*32 + 8*half + 16 + (j-8)
        const float4* hp = (const float4*)(hbase + kb * 32 + half * 8);
        float4 f0 = hp[0], f1 = hp[1];   // 8 floats at +0
        float4 f2 = hp[4], f3 = hp[5];   // 8 floats at +16
        v16bf a;
        a[0]  = (__bf16)f0.x; a[1]  = (__bf16)f0.y; a[2]  = (__bf16)f0.z; a[3]  = (__bf16)f0.w;
        a[4]  = (__bf16)f1.x; a[5]  = (__bf16)f1.y; a[6]  = (__bf16)f1.z; a[7]  = (__bf16)f1.w;
        a[8]  = (__bf16)f2.x; a[9]  = (__bf16)f2.y; a[10] = (__bf16)f2.z; a[11] = (__bf16)f2.w;
        a[12] = (__bf16)f3.x; a[13] = (__bf16)f3.y; a[14] = (__bf16)f3.z; a[15] = (__bf16)f3.w;
        #pragma unroll
        for (int ct = 0; ct < 8; ++ct) {
            // B elem j: K = kb*32 + 8*half + j (j<8), kb*32 + 8*half + 16 + (j-8) (j>=8), N = ct*16 + l16
            const v8bf* bp = (const v8bf*)(sW1t + (ct * 16 + l16) * LDP + kb * 32 + half * 8);
            v8bf blo = bp[0], bhi = bp[2];
            v16bf b;
            #pragma unroll
            for (int j = 0; j < 8; ++j) { b[j] = blo[j]; b[j + 8] = bhi[j]; }
            acc[ct] = __builtin_amdgcn_wmma_f32_16x16x32_bf16(
                false, a, false, b, (short)0, acc[ct], false, false);
        }
    }

    // ---- bias + ReLU, stage Y in LDS (A-fragment friendly row-major) ----
    #pragma unroll
    for (int ct = 0; ct < 8; ++ct) {
        float bias = b1[ct * 16 + l16];
        #pragma unroll
        for (int r = 0; r < 8; ++r) {                // D elem r -> row M = r + 8*half
            float y = acc[ct][r] + bias;
            y = y > 0.0f ? y : 0.0f;
            myY[(r + 8 * half) * LDP + ct * 16 + l16] = (__bf16)y;
        }
    }
    __syncthreads();   // cross-lane LDS store->load visibility

    // ---- GEMM2: acc = A(Y) @ W2 ----
    #pragma unroll
    for (int ct = 0; ct < 8; ++ct) acc[ct] = vzero;
    #pragma unroll
    for (int kb = 0; kb < 4; ++kb) {
        v16bf a;
        const __bf16* yrow = myY + l16 * LDP + kb * 32 + half * 8;
        const v8bf* ap0 = (const v8bf*)(yrow);        // j=0..7
        const v8bf* ap1 = (const v8bf*)(yrow + 16);   // j=8..15
        v8bf alo = ap0[0], ahi = ap1[0];
        #pragma unroll
        for (int j = 0; j < 8; ++j) { a[j] = alo[j]; a[j + 8] = ahi[j]; }
        #pragma unroll
        for (int ct = 0; ct < 8; ++ct) {
            const v8bf* bp = (const v8bf*)(sW2t + (ct * 16 + l16) * LDP + kb * 32 + half * 8);
            v8bf blo = bp[0], bhi = bp[2];
            v16bf b;
            #pragma unroll
            for (int j = 0; j < 8; ++j) { b[j] = blo[j]; b[j + 8] = bhi[j]; }
            acc[ct] = __builtin_amdgcn_wmma_f32_16x16x32_bf16(
                false, a, false, b, (short)0, acc[ct], false, false);
        }
    }

    // ---- bias + store (coalesced 64B per half-wave) ----
    #pragma unroll
    for (int ct = 0; ct < 8; ++ct) {
        float bias = b2[ct * 16 + l16];
        #pragma unroll
        for (int r = 0; r < 8; ++r) {
            long long orow = rowBase + r + 8 * half;
            if (orow < nrows)
                hout[orow * HDIM + ct * 16 + l16] = acc[ct][r] + bias;
        }
    }
}

// ---------------- pooling -----------------------------------------------------
__global__ void pool_zero(float* out, float* counts, int G) {
    int t = blockIdx.x * 256 + threadIdx.x;
    if (t < G * HDIM) out[t] = 0.0f;
    if (t < G) counts[t] = 0.0f;
}

__global__ void pool_acc(float* __restrict__ out, float* __restrict__ counts,
                         const float* __restrict__ h, const int* __restrict__ batch, int nrows) {
    long long tid = (long long)blockIdx.x * 256 + threadIdx.x;
    long long row = tid >> 5;
    if (row >= nrows) return;
    int lane = (int)(tid & 31);
    int b = batch[row];
    int c = lane << 2;
    const float4 v = *(const float4*)(h + row * HDIM + c);
    float* o = out + (long long)b * HDIM + c;
    atomicAdd(o + 0, v.x);
    atomicAdd(o + 1, v.y);
    atomicAdd(o + 2, v.z);
    atomicAdd(o + 3, v.w);
    if (lane == 0) atomicAdd(counts + b, 1.0f);
}

__global__ void pool_div(float* out, const float* __restrict__ counts, int G) {
    int t = blockIdx.x * 256 + threadIdx.x;
    if (t < G * HDIM) out[t] /= fmaxf(counts[t >> 7], 1.0f);
}

// ---------------- driver ------------------------------------------------------
extern "C" void kernel_launch(void* const* d_in, const int* in_sizes, int n_in,
                              void* d_out, int out_size, void* d_ws, size_t ws_size,
                              hipStream_t stream) {
    (void)n_in; (void)ws_size;
    const int N = in_sizes[0] / HDIM;       // 100000
    const int E = in_sizes[1] / 2;          // 1600000
    const int G = out_size / HDIM;          // 128

    const float* x   = (const float*)d_in[0];
    const int* src   = (const int*)d_in[1];
    const int* dst   = src + E;
    const int* batch = (const int*)d_in[2];

    float* bufA   = (float*)d_ws;                       // layer outputs  (N*128 f32)
    float* bufB   = bufA + (size_t)N * HDIM;            // aggregation    (N*128 f32)
    float* counts = bufB + (size_t)N * HDIM;            // G floats
    float* out    = (float*)d_out;

    const int n4 = N * HDIM / 4;
    const long long scatterThreads = (long long)E * 32;
    const unsigned scatterBlocks = (unsigned)((scatterThreads + 255) / 256);
    const unsigned mlpBlocks = (unsigned)((N + 127) / 128);

    const float* Hin = x;
    for (int l = 0; l < 3; ++l) {
        const float* W1 = (const float*)d_in[3 + l * 4 + 0];
        const float* b1 = (const float*)d_in[3 + l * 4 + 1];
        const float* W2 = (const float*)d_in[3 + l * 4 + 2];
        const float* b2 = (const float*)d_in[3 + l * 4 + 3];

        copy_kernel<<<(n4 + 255) / 256, 256, 0, stream>>>(bufB, Hin, n4);
        scatter_kernel<<<scatterBlocks, 256, 0, stream>>>(bufB, Hin, src, dst, E);
        mlp_kernel<<<mlpBlocks, 256, 0, stream>>>(bufA, bufB, W1, b1, W2, b2, N);
        Hin = bufA;
    }

    pool_zero<<<(G * HDIM + 255) / 256, 256, 0, stream>>>(out, counts, G);
    const long long poolThreads = (long long)N * 32;
    pool_acc<<<(unsigned)((poolThreads + 255) / 256), 256, 0, stream>>>(out, counts, bufA, batch, N);
    pool_div<<<(G * HDIM + 255) / 256, 256, 0, stream>>>(out, counts, G);
}
